// FullAttention_7275674599988
// MI455X (gfx1250) — matrix operand order repaired and checked by reading, not compiled, not run on hardware
//
#include <hip/hip_runtime.h>

// ---------------------------------------------------------------------------
// Causal multi-head attention (flash-attention style) for MI455X / gfx1250.
// B=2, L=S=2048, H=16, E=D=64. fp32 in/out, f16 WMMA compute, f32 accum.
// Double-buffered LDS staging; masking only on the diagonal KV tile.
// ---------------------------------------------------------------------------

typedef __attribute__((ext_vector_type(16))) _Float16 v16h;
typedef __attribute__((ext_vector_type(8)))  _Float16 v8h;
typedef __attribute__((ext_vector_type(8)))  float    v8f;
typedef __attribute__((ext_vector_type(4)))  float    v4f;

#define NB  2
#define NL  2048
#define NS  2048
#define NH  16
#define NE  64
#define ND  64

#define BM  128              // query rows per workgroup (8 waves x 16)
#define BN  64               // kv keys per tile
#define NWAVE 8

#define ATT_SCALE 0.125f     // 1/sqrt(64)
#define LOG2E     1.44269504f
#define NEG_BIG  (-3.0e38f)

__device__ __forceinline__ v8f wmma_f16(v16h a, v16h b, v8f c) {
    return __builtin_amdgcn_wmma_f32_16x16x32_f16(
        /*neg_a=*/false, a, /*neg_b=*/false, b,
        /*c_mod=*/(short)0, c, /*reuse_a=*/false, /*reuse_b=*/false);
}

// Coalesced global fetch of one thread's share of a K/V tile (4+4 float4).
__device__ __forceinline__ void tile_fetch(const float* kp, const float* vp,
                                           v4f kr[4], v4f vr[4]) {
    #pragma unroll
    for (int i = 0; i < 4; ++i) {
        kr[i] = *(const v4f*)(kp + 4 * i);
        vr[i] = *(const v4f*)(vp + 4 * i);
    }
}

// Convert to f16 and store into the LDS buffers (K row-major, V transposed).
__device__ __forceinline__ void tile_commit(_Float16 (*__restrict__ Kl)[NE],
                                            _Float16 (*__restrict__ Vt)[BN],
                                            int key, int eo,
                                            const v4f kr[4], const v4f vr[4]) {
    v8h t0, t1;
    #pragma unroll
    for (int i = 0; i < 4; ++i) {
        t0[i] = (_Float16)kr[0][i]; t0[i + 4] = (_Float16)kr[1][i];
        t1[i] = (_Float16)kr[2][i]; t1[i + 4] = (_Float16)kr[3][i];
    }
    *(v8h*)&Kl[key][eo]     = t0;                 // ds_store_b128
    *(v8h*)&Kl[key][eo + 8] = t1;
    #pragma unroll
    for (int j = 0; j < 4; ++j)
        #pragma unroll
        for (int i = 0; i < 4; ++i)
            Vt[eo + 4 * j + i][key] = (_Float16)vr[j][i];   // transposed scatter
}

__launch_bounds__(256, 2)
__global__ void fattn_causal_kernel(const float* __restrict__ Q,
                                    const float* __restrict__ K,
                                    const float* __restrict__ V,
                                    float* __restrict__ O) {
    // Double-buffered K/V staging + per-wave P slab.
    __shared__ __align__(128) _Float16 Kl[2][BN][NE];     // 16 KB
    __shared__ __align__(128) _Float16 Vt[2][ND][BN];     // 16 KB
    __shared__ __align__(128) _Float16 Pl[NWAVE][16][BN]; // 16 KB

    const int tid  = threadIdx.x;
    const int lane = tid & 31;
    const int wave = tid >> 5;
    const int ln16 = lane & 15;
    const int hf   = lane >> 4;          // 0: lanes 0-15, 1: lanes 16-31

    const int bh = blockIdx.y;           // b*NH + h
    const int b  = bh / NH;
    const int h  = bh % NH;
    const int q0 = blockIdx.x * BM;      // block's first query row
    const int qrow = q0 + wave * 16;     // this wave's first query row

    // ---- Q A-fragments (held for the whole pass). Contiguous 16-bit A
    // order: lane<16 -> Khat {0..15}; lane>=16 -> {16..31}.
    // Fragment qa0 covers E[0..31], qa1 covers E[32..63].
    v16h qa0, qa1;
    {
        const float* qp = Q + (((size_t)b * NL + (qrow + ln16)) * NH + h) * NE;
        const int k0 = hf * 16;
        v4f f0 = *(const v4f*)(qp + k0);
        v4f f1 = *(const v4f*)(qp + k0 + 4);
        v4f f2 = *(const v4f*)(qp + k0 + 8);
        v4f f3 = *(const v4f*)(qp + k0 + 12);
        v4f f4 = *(const v4f*)(qp + 32 + k0);
        v4f f5 = *(const v4f*)(qp + 32 + k0 + 4);
        v4f f6 = *(const v4f*)(qp + 32 + k0 + 8);
        v4f f7 = *(const v4f*)(qp + 32 + k0 + 12);
        #pragma unroll
        for (int i = 0; i < 4; ++i) {
            qa0[i]      = (_Float16)f0[i];  qa0[i + 4]  = (_Float16)f1[i];
            qa0[i + 8]  = (_Float16)f2[i];  qa0[i + 12] = (_Float16)f3[i];
            qa1[i]      = (_Float16)f4[i];  qa1[i + 4]  = (_Float16)f5[i];
            qa1[i + 8]  = (_Float16)f6[i];  qa1[i + 12] = (_Float16)f7[i];
        }
    }

    // Flash-attention running state. 8 row-slots per lane, matching the
    // 16x16 f32 C-layout (slot r <-> row qrow + r + hf*8).
    float m_run[8], l_run[8];
    v8f acc[4];
    #pragma unroll
    for (int r = 0; r < 8; ++r) { m_run[r] = NEG_BIG; l_run[r] = 0.0f; }
    #pragma unroll
    for (int t = 0; t < 4; ++t) acc[t] = (v8f){0,0,0,0,0,0,0,0};

    const int kend = (q0 + BM < NS) ? (q0 + BM) : NS;   // causal upper bound

    // Staging coordinates for this thread.
    const int skey = tid >> 2;              // 0..63
    const int seo  = (tid & 3) * 16;        // 0,16,32,48
    const float* kbase = K + (((size_t)b * NS + skey) * NH + h) * NE + seo;
    const float* vbase = V + (((size_t)b * NS + skey) * NH + h) * ND + seo;
    const size_t kvstep = (size_t)BN * NH * NE;   // NE==ND

    // ---- prologue: stage tile 0 into buffer 0 ----
    {
        v4f kr[4], vr[4];
        tile_fetch(kbase, vbase, kr, vr);
        tile_commit(Kl[0], Vt[0], skey, seo, kr, vr);
    }

    int cur = 0;
    for (int kb0 = 0; kb0 < kend; kb0 += BN) {
        __syncthreads();   // buffer `cur` fully staged; buffer cur^1 free

        // ---- software pipeline: issue next tile's global loads now ----
        const bool has_next = (kb0 + BN) < kend;     // block-uniform
        v4f kr[4], vr[4];
        if (has_next) {
            const float* kp = kbase + (size_t)(kb0 + BN) / BN * kvstep;
            const float* vp = vbase + (size_t)(kb0 + BN) / BN * kvstep;
            tile_fetch(kp, vp, kr, vr);
            if (kb0 + 2 * BN < kend) {               // prefetch tile i+2
                __builtin_prefetch(kp + kvstep, 0, 0);
                __builtin_prefetch(vp + kvstep, 0, 0);
            }
        }

        // Wave-uniform skip of fully-masked tiles (rows qrow..qrow+15 < kb0).
        if (kb0 <= qrow + 15) {
            // ---- S = Q K^T : 4 column tiles x 2 K-steps of WMMA ----
            v8f s[4];
            #pragma unroll
            for (int t = 0; t < 4; ++t) {
                // B-operand: lane = key column (t*16+ln16), 16 contiguous
                // Khat (=E) values starting at hf*16.
                v16h b0 = *(const v16h*)&Kl[cur][t * 16 + ln16][hf * 16];
                v16h b1 = *(const v16h*)&Kl[cur][t * 16 + ln16][32 + hf * 16];
                v8f c = (v8f){0,0,0,0,0,0,0,0};
                c = wmma_f16(qa0, b0, c);
                c = wmma_f16(qa1, b1, c);
                s[t] = c;
            }

            // ---- causal mask (diagonal tile only) + row max ----
            float mloc[8];
            #pragma unroll
            for (int r = 0; r < 8; ++r) mloc[r] = NEG_BIG;
            if (kb0 + BN - 1 > qrow) {               // wave-uniform: diag tile
                #pragma unroll
                for (int t = 0; t < 4; ++t) {
                    #pragma unroll
                    for (int r = 0; r < 8; ++r) {
                        const int row = qrow + r + hf * 8;
                        const int col = kb0 + t * 16 + ln16;
                        float v = s[t][r];
                        if (col > row) v = NEG_BIG;
                        s[t][r] = v;
                        mloc[r] = fmaxf(mloc[r], v);
                    }
                }
            } else {                                  // interior: no compares
                #pragma unroll
                for (int t = 0; t < 4; ++t)
                    #pragma unroll
                    for (int r = 0; r < 8; ++r)
                        mloc[r] = fmaxf(mloc[r], s[t][r]);
            }
            #pragma unroll
            for (int off = 1; off < 16; off <<= 1) {
                #pragma unroll
                for (int r = 0; r < 8; ++r)
                    mloc[r] = fmaxf(mloc[r], __shfl_xor(mloc[r], off, 32));
            }

            // ---- online softmax update (base-2 exponentials) ----
            const float c2 = ATT_SCALE * LOG2E;
            float alpha[8], mnew[8], rs[8];
            #pragma unroll
            for (int r = 0; r < 8; ++r) {
                mnew[r]  = fmaxf(m_run[r], mloc[r]);
                alpha[r] = exp2f((m_run[r] - mnew[r]) * c2);
                rs[r]    = 0.0f;
            }
            #pragma unroll
            for (int t = 0; t < 4; ++t) {
                #pragma unroll
                for (int r = 0; r < 8; ++r) {
                    float p = exp2f((s[t][r] - mnew[r]) * c2);
                    s[t][r] = p;
                    rs[r] += p;
                }
            }
            #pragma unroll
            for (int off = 1; off < 16; off <<= 1) {
                #pragma unroll
                for (int r = 0; r < 8; ++r)
                    rs[r] += __shfl_xor(rs[r], off, 32);
            }
            #pragma unroll
            for (int r = 0; r < 8; ++r) {
                l_run[r] = l_run[r] * alpha[r] + rs[r];
                m_run[r] = mnew[r];
            }
            #pragma unroll
            for (int t = 0; t < 4; ++t)
                #pragma unroll
                for (int r = 0; r < 8; ++r)
                    acc[t][r] *= alpha[r];

            // ---- re-layout P: C-layout -> LDS -> A-layout fragments ----
            #pragma unroll
            for (int t = 0; t < 4; ++t)
                #pragma unroll
                for (int r = 0; r < 8; ++r)
                    Pl[wave][r + hf * 8][t * 16 + ln16] = (_Float16)s[t][r];
            asm volatile("s_wait_dscnt 0x0" ::: "memory");  // own-wave DS RAW

            const _Float16* prow = &Pl[wave][ln16][0];
            const int k0 = hf * 16;
            v8h pl0 = *(const v8h*)(prow + k0);
            v8h ph0 = *(const v8h*)(prow + k0 + 8);
            v8h pl1 = *(const v8h*)(prow + 32 + k0);
            v8h ph1 = *(const v8h*)(prow + 32 + k0 + 8);
            v16h pa0 = __builtin_shufflevector(pl0, ph0,
                0,1,2,3,4,5,6,7,8,9,10,11,12,13,14,15);
            v16h pa1 = __builtin_shufflevector(pl1, ph1,
                0,1,2,3,4,5,6,7,8,9,10,11,12,13,14,15);

            // ---- O += P V : 4 d-tiles x 2 key-steps of WMMA ----
            #pragma unroll
            for (int t = 0; t < 4; ++t) {
                v16h vb0 = *(const v16h*)&Vt[cur][t * 16 + ln16][hf * 16];
                v16h vb1 = *(const v16h*)&Vt[cur][t * 16 + ln16][32 + hf * 16];
                acc[t] = wmma_f16(pa0, vb0, acc[t]);
                acc[t] = wmma_f16(pa1, vb1, acc[t]);
            }
        }

        // ---- commit next tile into the alternate buffer ----
        if (has_next)
            tile_commit(Kl[cur ^ 1], Vt[cur ^ 1], skey, seo, kr, vr);
        cur ^= 1;
    }

    // ---- epilogue: normalize and store (coalesced b32 per 16 lanes) ----
    float inv_l[8];
    #pragma unroll
    for (int r = 0; r < 8; ++r)
        inv_l[r] = (l_run[r] > 0.0f) ? (1.0f / l_run[r]) : 0.0f;
    #pragma unroll
    for (int t = 0; t < 4; ++t) {
        #pragma unroll
        for (int r = 0; r < 8; ++r) {
            const int row = qrow + r + hf * 8;
            const int col = t * 16 + ln16;
            O[(((size_t)b * NL + row) * NH + h) * ND + col] = acc[t][r] * inv_l[r];
        }
    }
}

extern "C" void kernel_launch(void* const* d_in, const int* in_sizes, int n_in,
                              void* d_out, int out_size, void* d_ws, size_t ws_size,
                              hipStream_t stream) {
    const float* Q = (const float*)d_in[0];   // [B,L,H,E]
    const float* K = (const float*)d_in[1];   // [B,S,H,E]
    const float* V = (const float*)d_in[2];   // [B,S,H,D]
    // d_in[3] = attn_mask (bool) — causal, synthesized in-kernel.
    float* O = (float*)d_out;                 // [B,L,H,D]
    dim3 grid(NL / BM, NB * NH);
    fattn_causal_kernel<<<grid, 256, 0, stream>>>(Q, K, V, O);
}
